// Head_60361470378484
// MI455X (gfx1250) — matrix operand order repaired and checked by reading, not compiled, not run on hardware
//
#include <hip/hip_runtime.h>
#include <hip/hip_bf16.h>

typedef __attribute__((ext_vector_type(16))) _Float16 v16h;
typedef __attribute__((ext_vector_type(8)))  _Float16 v8h;
typedef __attribute__((ext_vector_type(8)))  float    v8f;

#define TSEQ 4096
#define NB   4
#define CDIM 1024
#define HS   64

static __device__ __forceinline__ v8f wmma_f16(v16h a, v16h b, v8f c) {
  // D = A(16x32 f16) * B(32x16 f16) + C(16x16 f32)
  return __builtin_amdgcn_wmma_f32_16x16x32_f16(false, a, false, b, (short)0, c,
                                                false, false);
}

static __device__ __forceinline__ v16h cat16(v8h lo, v8h hi) {
  v16h r;
#pragma unroll
  for (int i = 0; i < 8; ++i) { r[i] = lo[i]; r[i + 8] = hi[i]; }
  return r;
}

// ---------------------------------------------------------------------------
// Kernel 1: QKV projection.  x[16384,1024]f32 x W[1024,64]f32 (x3) -> f16.
// Reference swaps names: q = x@Wk, k = x@Wq, v = x@Wv.
// Outputs: Qh (pre-scaled by 1/32), Kh row-major [B*T,64] f16 ;
//          Vt transposed [B][64][T] f16.
// Per block: 8 waves x 16 rows = 128 rows.  W K-chunk staged to LDS
// transposed+f16; B fragments preloaded in groups of 4 simultaneously-live
// registers so ds_load latency overlaps the XDL pipe (staggered dscnt waits
// instead of s_wait_dscnt 0 before every wmma).
// ---------------------------------------------------------------------------
__global__ __launch_bounds__(256) void proj_qkv(
    const float* __restrict__ x,
    const float* __restrict__ Wq, const float* __restrict__ Wk,
    const float* __restrict__ Wv,
    _Float16* __restrict__ Qh, _Float16* __restrict__ Kh,
    _Float16* __restrict__ Vt)
{
  __shared__ __align__(64) _Float16 lw[192][32];   // [outCol][k] transposed chunk
  const int tid  = threadIdx.x;
  const int lane = tid & 31;
  const int wave = tid >> 5;
  const int hlf  = lane >> 4;        // half-wave: 0 or 1
  const int l16  = lane & 15;
  const int rowBase = blockIdx.x * 128 + wave * 16;
  const int kbA = hlf ? 16 : 0;      // A-frag K-base (contiguous K order, matches B)
  const int kbB = hlf ? 16 : 0;      // B-frag K-base
  const int arow = rowBase + l16;

  const v8f vzero = {0.f, 0.f, 0.f, 0.f, 0.f, 0.f, 0.f, 0.f};
  v8f acc[12];
#pragma unroll
  for (int i = 0; i < 12; ++i) acc[i] = vzero;

  for (int kc = 0; kc < CDIM / 32; ++kc) {
    __syncthreads();                       // protect prior-iteration reads
    for (int idx = tid; idx < 192 * 32; idx += 256) {
      const int r = idx / 192;             // k within chunk (0..31)
      const int c = idx - r * 192;         // output col (0..191)
      const float* Ws = (c < 64) ? Wk : ((c < 128) ? Wq : Wv);
      lw[c][r] = (_Float16)Ws[(size_t)(kc * 32 + r) * HS + (c & 63)];
    }
    __syncthreads();

    // A fragment from x (f32 -> f16): row = arow, K = kbA+{0..15}
    const float* xr = x + (size_t)arow * CDIM + kc * 32 + kbA;
    if (kc + 1 < CDIM / 32) __builtin_prefetch(xr + 32, 0, 0);  // next K-chunk
    v16h af;
#pragma unroll
    for (int i = 0; i < 8; ++i) {
      af[i]     = (_Float16)xr[i];
      af[i + 8] = (_Float16)xr[i + 8];
    }

    // 4 simultaneously-live B fragments per group: 8 ds_load_b128 issued
    // back-to-back, wmmas retire with staggered dscnt waits.
#pragma unroll
    for (int g = 0; g < 3; ++g) {
      const v16h b0 = *(const v16h*)&lw[(g * 4 + 0) * 16 + l16][kbB];
      const v16h b1 = *(const v16h*)&lw[(g * 4 + 1) * 16 + l16][kbB];
      const v16h b2 = *(const v16h*)&lw[(g * 4 + 2) * 16 + l16][kbB];
      const v16h b3 = *(const v16h*)&lw[(g * 4 + 3) * 16 + l16][kbB];
      acc[g * 4 + 0] = wmma_f16(af, b0, acc[g * 4 + 0]);
      acc[g * 4 + 1] = wmma_f16(af, b1, acc[g * 4 + 1]);
      acc[g * 4 + 2] = wmma_f16(af, b2, acc[g * 4 + 2]);
      acc[g * 4 + 3] = wmma_f16(af, b3, acc[g * 4 + 3]);
    }
  }

  // D layout: row m = j + 8*hlf, col n = l16 (per 16-wide tile)
#pragma unroll
  for (int nt = 0; nt < 12; ++nt) {
    const int c = nt * 16 + l16;
#pragma unroll
    for (int j = 0; j < 8; ++j) {
      const int gr = rowBase + j + 8 * hlf;
      if (c < 64) {
        // fold softmax scale C^-0.5 = 1/32 (exact power-of-two in f16)
        Qh[(size_t)gr * HS + c] = (_Float16)(acc[nt][j] * 0.03125f);
      } else if (c < 128) {
        Kh[(size_t)gr * HS + (c - 64)] = (_Float16)acc[nt][j];
      } else {
        const int bb = gr >> 12, tt = gr & (TSEQ - 1);
        Vt[((size_t)bb * HS + (c - 128)) * TSEQ + tt] = (_Float16)acc[nt][j];
      }
    }
  }
}

// ---------------------------------------------------------------------------
// One 32-key flash-attention tile.  MASK=true only for the single tile that
// straddles the causal diagonal (all earlier tiles are fully unmasked since
// k0 <= q0 for every processed tile).
// ---------------------------------------------------------------------------
template <bool MASK>
static __device__ __forceinline__ void attn_tile(
    int k0, int b, int q0, int hlf, int l16, int kbA, int kbB,
    const _Float16* __restrict__ Kh, const _Float16* __restrict__ Vt,
    const v16h (&qf)[2], v8f (&o)[4], float (&m)[8], float (&l)[8],
    _Float16 (&pw)[16][32])
{
  const v8f vzero = {0.f, 0.f, 0.f, 0.f, 0.f, 0.f, 0.f, 0.f};

  // Preload all 4 K fragments (simultaneously live -> loads pipeline)
  const _Float16* kbase = Kh + ((size_t)b * TSEQ + k0) * HS + kbB;
  const v16h kf00 = *(const v16h*)(kbase + (size_t)l16 * HS);            // t=0,hc=0
  const v16h kf01 = *(const v16h*)(kbase + (size_t)l16 * HS + 32);       // t=0,hc=1
  const v16h kf10 = *(const v16h*)(kbase + (size_t)(16 + l16) * HS);     // t=1,hc=0
  const v16h kf11 = *(const v16h*)(kbase + (size_t)(16 + l16) * HS + 32);// t=1,hc=1

  v8f s0 = wmma_f16(qf[0], kf00, vzero);
  v8f s1 = wmma_f16(qf[0], kf10, vzero);
  s0 = wmma_f16(qf[1], kf01, s0);
  s1 = wmma_f16(qf[1], kf11, s1);

  float sv[2][8];
#pragma unroll
  for (int t = 0; t < 2; ++t) {
    const int key = k0 + t * 16 + l16;
#pragma unroll
    for (int j = 0; j < 8; ++j) {
      const float v = (t == 0) ? s0[j] : s1[j];
      if (MASK) {
        const int qr = q0 + j + 8 * hlf;
        sv[t][j] = (key <= qr) ? v : -__builtin_inff();
      } else {
        sv[t][j] = v;
      }
    }
  }

  // Row max across the 16 lanes of this half-wave (row-uniform after xor tree)
  float rm[8];
#pragma unroll
  for (int j = 0; j < 8; ++j) rm[j] = fmaxf(sv[0][j], sv[1][j]);
#pragma unroll
  for (int off = 1; off < 16; off <<= 1)
#pragma unroll
    for (int j = 0; j < 8; ++j) rm[j] = fmaxf(rm[j], __shfl_xor(rm[j], off, 32));

  // Online softmax update; l kept as per-lane partial (same rescale chain)
#pragma unroll
  for (int j = 0; j < 8; ++j) {
    const float mn = fmaxf(m[j], rm[j]);
    const float al = __expf(m[j] - mn);
    m[j] = mn;
    l[j] *= al;
#pragma unroll
    for (int nt = 0; nt < 4; ++nt) o[nt][j] *= al;
#pragma unroll
    for (int t = 0; t < 2; ++t) {
      const float p = __expf(sv[t][j] - mn);
      l[j] += p;
      pw[j + 8 * hlf][t * 16 + l16] = (_Float16)p;  // D-layout -> LDS
    }
  }
  asm volatile("" ::: "memory");  // keep LDS store->load order (same-wave DS is in-order)

  // Re-read P as A fragment (16x32), then O += P.V  (V is pre-transposed).
  // All 4 V fragments preloaded so global loads pipeline ahead of the wmmas.
  const _Float16* pp = &pw[l16][kbA];
  const v16h pf = cat16(*(const v8h*)pp, *(const v8h*)(pp + 8));
  const _Float16* vbase = Vt + ((size_t)b * HS + l16) * TSEQ + k0 + kbB;
  const v16h vf0 = *(const v16h*)(vbase);
  const v16h vf1 = *(const v16h*)(vbase + (size_t)16 * TSEQ);
  const v16h vf2 = *(const v16h*)(vbase + (size_t)32 * TSEQ);
  const v16h vf3 = *(const v16h*)(vbase + (size_t)48 * TSEQ);
  o[0] = wmma_f16(pf, vf0, o[0]);
  o[1] = wmma_f16(pf, vf1, o[1]);
  o[2] = wmma_f16(pf, vf2, o[2]);
  o[3] = wmma_f16(pf, vf3, o[3]);
}

// ---------------------------------------------------------------------------
// Kernel 2: causal flash attention.  One wave per 16-query tile, streams
// 32-key tiles with online softmax; A never materialized in HBM.
// ---------------------------------------------------------------------------
__global__ __launch_bounds__(128) void attn_flash(
    const _Float16* __restrict__ Qh, const _Float16* __restrict__ Kh,
    const _Float16* __restrict__ Vt, float* __restrict__ out)
{
  __shared__ __align__(64) _Float16 pbuf[4][16][32];  // per-wave P tile
  const int tid  = threadIdx.x;
  const int lane = tid & 31;
  const int wave = tid >> 5;
  const int hlf  = lane >> 4;
  const int l16  = lane & 15;
  const int kbA  = hlf ? 16 : 0;
  const int kbB  = hlf ? 16 : 0;

  const int b  = blockIdx.x / (TSEQ / 64);
  const int q0 = (blockIdx.x % (TSEQ / 64)) * 64 + wave * 16;
  const size_t rowg = (size_t)b * TSEQ + q0;

  // Preload the 16x64 query tile as two 16x32 A fragments (loop-invariant).
  v16h qf[2];
#pragma unroll
  for (int hc = 0; hc < 2; ++hc) {
    const _Float16* qp = Qh + (rowg + l16) * HS + hc * 32 + kbA;
    qf[hc] = cat16(*(const v8h*)qp, *(const v8h*)(qp + 8));
  }

  const v8f vzero = {0.f, 0.f, 0.f, 0.f, 0.f, 0.f, 0.f, 0.f};
  v8f o[4];
#pragma unroll
  for (int nt = 0; nt < 4; ++nt) o[nt] = vzero;
  float m[8], l[8];
#pragma unroll
  for (int j = 0; j < 8; ++j) { m[j] = -__builtin_inff(); l[j] = 0.f; }

  const int ntiles = (q0 + 47) >> 5;   // causal: keys <= q0+15

  // Fast path: fully-unmasked tiles (k0+31 <= q0-1 for all but the last tile)
  for (int kt = 0; kt + 1 < ntiles; ++kt) {
    // prefetch next tile's K rows (each lane one key row => whole 32x64 tile)
    const int kn = (kt + 1) * 32;
    __builtin_prefetch(Kh + ((size_t)b * TSEQ + kn + lane) * HS, 0, 0);
    __builtin_prefetch(Vt + ((size_t)b * HS + lane) * TSEQ + kn, 0, 0);
    __builtin_prefetch(Vt + ((size_t)b * HS + 32 + lane) * TSEQ + kn, 0, 0);
    attn_tile<false>(kt * 32, b, q0, hlf, l16, kbA, kbB, Kh, Vt,
                     qf, o, m, l, pbuf[wave]);
  }
  // Diagonal tile (the only one that can be partially masked)
  attn_tile<true>((ntiles - 1) * 32, b, q0, hlf, l16, kbA, kbB, Kh, Vt,
                  qf, o, m, l, pbuf[wave]);

  // Reduce the partial row sums across the half-wave, normalize, store f32
#pragma unroll
  for (int off = 1; off < 16; off <<= 1)
#pragma unroll
    for (int j = 0; j < 8; ++j) l[j] += __shfl_xor(l[j], off, 32);

#pragma unroll
  for (int nt = 0; nt < 4; ++nt)
#pragma unroll
    for (int j = 0; j < 8; ++j) {
      const int qr = q0 + j + 8 * hlf;
      out[((size_t)b * TSEQ + qr) * HS + nt * 16 + l16] = o[nt][j] / l[j];
    }
}

// ---------------------------------------------------------------------------
extern "C" void kernel_launch(void* const* d_in, const int* in_sizes, int n_in,
                              void* d_out, int out_size, void* d_ws, size_t ws_size,
                              hipStream_t stream) {
  (void)in_sizes; (void)n_in; (void)out_size; (void)ws_size;
  const float* x  = (const float*)d_in[0];
  const float* Wq = (const float*)d_in[1];
  const float* Wk = (const float*)d_in[2];
  const float* Wv = (const float*)d_in[3];

  // Workspace: Qh | Kh | Vt  (f16), 3 * 16384*64 halves = 6 MB
  _Float16* Qh = (_Float16*)d_ws;
  _Float16* Kh = Qh + (size_t)NB * TSEQ * HS;
  _Float16* Vt = Kh + (size_t)NB * TSEQ * HS;

  proj_qkv<<<dim3(NB * TSEQ / 128), dim3(256), 0, stream>>>(x, Wq, Wk, Wv,
                                                            Qh, Kh, Vt);
  attn_flash<<<dim3(NB * (TSEQ / 64)), dim3(128), 0, stream>>>(Qh, Kh, Vt,
                                                               (float*)d_out);
}
